// InteractionPPBlock_11940009083127
// MI455X (gfx1250) — matrix operand order repaired and checked by reading, not compiled, not run on hardware
//
#include <hip/hip_runtime.h>
#include <hip/hip_bf16.h>

// ---------------------------------------------------------------------------
// DimeNet++ InteractionPPBlock for MI455X (gfx1250, wave32, WMMA + TDM).
// - All dense GEMMs: v_wmma_f32_16x16x32_f16 (f16 in, f32 accumulate).
// - Weights converted + pre-swizzled once into B-fragment order, then each
//   8-wave workgroup stages its column strip of weights into LDS with the
//   Tensor Data Mover (tensor_load_to_lds + s_wait_tensorcnt) and all waves
//   read B fragments from LDS (ds_load_b128), amortizing weight traffic 8x.
// - Triplet gather/scatter runs on global_atomic_add_f32; x_kj/agg (67 MB
//   each) are L2-resident on the 192 MB L2.
// ---------------------------------------------------------------------------

typedef _Float16 half_t;
typedef _Float16 v16h __attribute__((ext_vector_type(16)));
typedef float    v8f  __attribute__((ext_vector_type(8)));
typedef unsigned int u32x4 __attribute__((ext_vector_type(4)));
typedef int          i32x4 __attribute__((ext_vector_type(4)));
typedef int          i32x8 __attribute__((ext_vector_type(8)));

#define E_EDGES 262144
#define T_TRIP  2097152
#define EMB     128
#define INTD    64
#define BAS     8
#define NRD     6
#define NSB     42   // NR*NS
#define ROWS_PER_BLK 8   // waves per workgroup == row tiles per workgroup

#if defined(__AMDGCN__) && __has_builtin(__builtin_amdgcn_tensor_load_to_lds) && \
    __has_builtin(__builtin_amdgcn_s_wait_tensorcnt)
#define HAS_TDM 1
#endif

__device__ __forceinline__ float silu_f(float x) {
    return x / (1.0f + __expf(-x));
}

// ---- A fragment: 16x32 f16 tile built from row-major f32 (contiguous k order, same as B packing) ----
// lanes 0-15 : row M=lane,   halves 0..15 = K 0..15
// lanes 16-31: row M=lane-16, halves 0..15 = K 16..31
__device__ __forceinline__ v16h load_a_f32(const float* __restrict__ A, int lda,
                                           int row0, int k0, int lane) {
    int r    = lane & 15;
    int ksel = (lane & 16) ? 16 : 0;
    const float* p = A + (size_t)(row0 + r) * lda + k0 + ksel;
    v16h a;
#pragma unroll
    for (int i = 0; i < 8; ++i) a[i] = (half_t)p[i];
#pragma unroll
    for (int i = 0; i < 8; ++i) a[8 + i] = (half_t)p[8 + i];
    return a;
}

// ---- B fragment from LDS-staged swizzled strip (16 contiguous halves/lane) --
__device__ __forceinline__ v16h load_b_lds(const half_t* s_b, int kt, int lane) {
    return *(const v16h*)(s_b + ((size_t)kt * 32 + lane) * 16);
}

// ---------------------------------------------------------------------------
// TDM: DMA a 2D tile (nrows x row_bytes, global stride row_stride_bytes) of
// 4-byte elements from global memory into LDS at byte offset lds_off.
// D# fields per CDNA5 ISA ch.8: group0 {count=1, lds_addr, global_addr,
// type=2}; group1 {data_size=4B, tensor_dim0/1, tile_dim0/1, dim0_stride}.
// Issued by one wave; completion via s_wait_tensorcnt 0.
// ---------------------------------------------------------------------------
__device__ __forceinline__ void tdm_load_2d(unsigned lds_off, const void* gsrc,
                                            unsigned row_bytes, unsigned nrows,
                                            unsigned row_stride_bytes) {
#ifdef HAS_TDM
    unsigned re = row_bytes >> 2;          // elements per tile row (4B elems)
    unsigned se = row_stride_bytes >> 2;   // row stride in elements
    unsigned long long ga = (unsigned long long)(size_t)gsrc;

    union { u32x4 v; unsigned u[4]; } g0 = {};
    g0.u[0] = 1u;                                  // count=1 (valid), user mode
    g0.u[1] = lds_off;                             // lds_addr (bytes)
    g0.u[2] = (unsigned)ga;                        // global_addr[31:0]
    g0.u[3] = (unsigned)((ga >> 32) & 0x1FFFFFFull) | (2u << 30); // [56:32]|type=2

    union { i32x8 v; unsigned long long q[4]; } g1 = {};
    g1.q[0] = (2ull << 16)                                   // data_size = 4B
            | ((unsigned long long)(se & 0xFFFFu) << 48);    // tensor_dim0 lo16
    g1.q[1] = ((unsigned long long)(se >> 16) & 0xFFFFull)   // tensor_dim0 hi16
            | ((unsigned long long)nrows << 16)              // tensor_dim1
            | ((unsigned long long)(re & 0xFFFFu) << 48);    // tile_dim0
    g1.q[2] = (unsigned long long)(nrows & 0xFFFFu)          // tile_dim1
            | ((unsigned long long)se << 32);                // dim0_stride lo32
    g1.q[3] = 0ull;                                          // stride hi + dim1_stride=0

    i32x4 z4 = {0, 0, 0, 0};
#if __clang_major__ >= 23
    i32x8 z8 = {0, 0, 0, 0, 0, 0, 0, 0};
    __builtin_amdgcn_tensor_load_to_lds(g0.v, g1.v, z4, z4, z8, 0);
#else
    __builtin_amdgcn_tensor_load_to_lds(g0.v, g1.v, z4, z4, 0);
#endif
#else
    (void)lds_off; (void)gsrc; (void)row_bytes; (void)nrows; (void)row_stride_bytes;
#endif
}

// ---------------------------------------------------------------------------
// Weight prep: f32 [K,N] row-major -> f16 in B-fragment order.
// B layout (32x16 K-tile): lanes 0-15 hold K=0..15 (halves in K order),
// lanes 16-31 hold K=16..31; lane%16 = column N within the 16-wide tile.
// Chunk (kt,nt) is 1KB contiguous at ((kt*nTiles+nt)*1024) bytes.
// ---------------------------------------------------------------------------
__global__ void k_swz_weights(const float* __restrict__ W, half_t* __restrict__ dst,
                              int K, int N) {
    int i = blockIdx.x * blockDim.x + threadIdx.x;
    if (i >= K * N) return;
    int k = i / N, n = i % N;
    int kt = k >> 5, kk = k & 31;
    int nt = n >> 4, nn = n & 15;
    int lane = nn + ((kk >> 4) << 4);
    int kk2  = kk & 15;
    size_t off = (((size_t)kt * (N >> 4) + nt) * 32 + lane) * 16 + kk2;
    dst[off] = (half_t)W[(size_t)k * N + n];
}

// Fuse W_rbf1[6,8] @ W_rbf2[8,128] -> Wrbf[6,128] (f32, tiny)
__global__ void k_fuse_rbf(const float* __restrict__ W1, const float* __restrict__ W2,
                           float* __restrict__ out) {
    int i = blockIdx.x * blockDim.x + threadIdx.x;
    if (i >= NRD * EMB) return;
    int r = i / EMB, n = i % EMB;
    float acc = 0.f;
#pragma unroll
    for (int b = 0; b < BAS; ++b) acc += W1[r * BAS + b] * W2[b * EMB + n];
    out[i] = acc;
}

__global__ void k_zero(float* __restrict__ p, size_t n) {
    size_t i = (size_t)blockIdx.x * blockDim.x + threadIdx.x;
    size_t s = (size_t)gridDim.x * blockDim.x;
    for (; i < n; i += s) p[i] = 0.f;
}

// ---------------------------------------------------------------------------
// Generic WMMA GEMM:  out[M,N] = silu(A[M,K]_f32 @ Bswz_f16 + bias) (+ add)
// grid = (M/128, N/16), block = 256 (8 waves; wave w owns row tile w).
// Column strip of B (kTiles x 1KB) staged into LDS via TDM by wave 0.
// ---------------------------------------------------------------------------
__global__ __launch_bounds__(256)
void k_gemm16(const float* __restrict__ A, int K, int N,
              const half_t* __restrict__ Bw,
              const float* __restrict__ bias,
              const float* __restrict__ addsrc,
              float* __restrict__ out) {
    __shared__ __align__(32) half_t s_b[4 * 512];   // up to 4 K-tiles (4 KB)
    int lane = threadIdx.x & 31;
    int wave = threadIdx.x >> 5;
    int nt   = blockIdx.y;
    int col0 = nt * 16;
    int nTiles = N >> 4;
    int kTiles = K >> 5;
    int row0 = (blockIdx.x * ROWS_PER_BLK + wave) * 16;
    int col  = col0 + (lane & 15);

#ifdef HAS_TDM
    if (threadIdx.x < 32) {   // wave 0 drives the Tensor Data Mover
        tdm_load_2d(0u, Bw + (size_t)nt * 512, 1024u, (unsigned)kTiles,
                    (unsigned)nTiles * 1024u);
        __builtin_amdgcn_s_wait_tensorcnt(0);
    }
#else
    for (int i = threadIdx.x; i < kTiles * 256; i += 256) {
        int kt = i >> 8, e = i & 255;
        ((unsigned*)s_b)[i] =
            ((const unsigned*)Bw)[((size_t)kt * nTiles + nt) * 256 + e];
    }
#endif
    __syncthreads();

    if (addsrc) __builtin_prefetch(addsrc + (size_t)(row0 + (lane & 15)) * N + col0, 0, 1);

    float bv = bias ? bias[col] : 0.f;
    v8f c;
#pragma unroll
    for (int i = 0; i < 8; ++i) c[i] = bv;

    for (int kt = 0; kt < kTiles; ++kt) {
        v16h a = load_a_f32(A, K, row0, kt * 32, lane);
        v16h b = load_b_lds(s_b, kt, lane);
        c = __builtin_amdgcn_wmma_f32_16x16x32_f16(false, a, false, b, (short)0, c,
                                                   false, false);
    }

    int rbase = row0 + ((lane & 16) ? 8 : 0);
#pragma unroll
    for (int r = 0; r < 8; ++r) {
        float v = silu_f(c[r]);
        size_t o = (size_t)(rbase + r) * N + col;
        if (addsrc) v += addsrc[o];
        out[o] = v;
    }
}

// ---------------------------------------------------------------------------
// Edge transfer: x_ji = silu(m@W_ji+b_ji); x_kjg = silu(m@W_kj+b_kj)*(rbf@Wrbf)
// Shares A fragments of m between both GEMMs (8 WMMA per wave).
// grid = (E/128, EMB/16), block = 256. Both weight strips TDM'd to LDS.
// ---------------------------------------------------------------------------
__global__ __launch_bounds__(256)
void k_edge_transfer(const float* __restrict__ m,
                     const half_t* __restrict__ Wji, const half_t* __restrict__ Wkj,
                     const float* __restrict__ bji, const float* __restrict__ bkj,
                     const float* __restrict__ rbf, const float* __restrict__ Wrbf,
                     float* __restrict__ xji, float* __restrict__ xkjg) {
    __shared__ __align__(32) half_t s_w[2][4 * 512];  // [0]=W_ji strip, [1]=W_kj
    __shared__ float s_rbf[16 * ROWS_PER_BLK][NRD];
    int lane = threadIdx.x & 31;
    int wave = threadIdx.x >> 5;
    int nt   = blockIdx.y;
    int col0 = nt * 16;
    int rowB = blockIdx.x * (16 * ROWS_PER_BLK);      // block's first edge
    int row0 = rowB + wave * 16;                      // this wave's row tile

#ifdef HAS_TDM
    if (threadIdx.x < 32) {
        tdm_load_2d(0u,    Wji + (size_t)nt * 512, 1024u, 4u, 8u * 1024u);
        tdm_load_2d(4096u, Wkj + (size_t)nt * 512, 1024u, 4u, 8u * 1024u);
        __builtin_amdgcn_s_wait_tensorcnt(0);
    }
#else
    for (int i = threadIdx.x; i < 4 * 256; i += 256) {
        int kt = i >> 8, e = i & 255;
        ((unsigned*)s_w[0])[i] = ((const unsigned*)Wji)[((size_t)kt * 8 + nt) * 256 + e];
        ((unsigned*)s_w[1])[i] = ((const unsigned*)Wkj)[((size_t)kt * 8 + nt) * 256 + e];
    }
#endif
    for (int i = threadIdx.x; i < 16 * ROWS_PER_BLK * NRD; i += 256)
        s_rbf[i / NRD][i % NRD] = rbf[(size_t)(rowB + i / NRD) * NRD + (i % NRD)];
    __syncthreads();

    int col = col0 + (lane & 15);
    float b1 = bji[col], b2 = bkj[col];
    v8f cji, ckj;
#pragma unroll
    for (int i = 0; i < 8; ++i) { cji[i] = b1; ckj[i] = b2; }

#pragma unroll
    for (int kt = 0; kt < 4; ++kt) {
        v16h a  = load_a_f32(m, EMB, row0, kt * 32, lane);
        v16h w1 = load_b_lds(s_w[0], kt, lane);
        v16h w2 = load_b_lds(s_w[1], kt, lane);
        cji = __builtin_amdgcn_wmma_f32_16x16x32_f16(false, a, false, w1, (short)0, cji,
                                                     false, false);
        ckj = __builtin_amdgcn_wmma_f32_16x16x32_f16(false, a, false, w2, (short)0, ckj,
                                                     false, false);
    }

    float wcol[NRD];
#pragma unroll
    for (int j = 0; j < NRD; ++j) wcol[j] = Wrbf[j * EMB + col];

    int rloc = (lane & 16) ? 8 : 0;
#pragma unroll
    for (int r = 0; r < 8; ++r) {
        int rl = rloc + r;
        float g = 0.f;
#pragma unroll
        for (int j = 0; j < NRD; ++j) g += s_rbf[wave * 16 + rl][j] * wcol[j];
        size_t o = (size_t)(row0 + rl) * EMB + col;
        xji[o]  = silu_f(cji[r]);
        xkjg[o] = silu_f(ckj[r]) * g;
    }
}

// ---------------------------------------------------------------------------
// Triplet message: sbf MLP (42->8->64), gate by gathered x_kj[idx_src],
// scatter-add into agg[idx_dst] with global_atomic_add_f32 (L2-resident).
// ---------------------------------------------------------------------------
__global__ __launch_bounds__(256)
void k_msg(const float* __restrict__ sbf,
           const int* __restrict__ idx_src, const int* __restrict__ idx_dst,
           const float* __restrict__ W1 /*42x8*/, const float* __restrict__ W2 /*8x64*/,
           const float* __restrict__ xk /*E x 64*/, float* __restrict__ agg /*E x 64*/,
           int T) {
    __shared__ float sW1[NSB * BAS];
    __shared__ float sW2[BAS * INTD];
    for (int i = threadIdx.x; i < NSB * BAS; i += 256) sW1[i] = W1[i];
    for (int i = threadIdx.x; i < BAS * INTD; i += 256) sW2[i] = W2[i];
    __syncthreads();

    int t = blockIdx.x * 256 + threadIdx.x;
    if (t >= T) return;

    float tmp[BAS];
#pragma unroll
    for (int j = 0; j < BAS; ++j) tmp[j] = 0.f;
    const float* sr = sbf + (size_t)t * NSB;
    for (int i = 0; i < NSB; ++i) {
        float s = sr[i];
#pragma unroll
        for (int j = 0; j < BAS; ++j) tmp[j] += s * sW1[i * BAS + j];
    }

    int src = idx_src[t], dst = idx_dst[t];
    const float* xr = xk + (size_t)src * INTD;
    float* ar = agg + (size_t)dst * INTD;
#pragma unroll 4
    for (int c = 0; c < INTD; ++c) {
        float s = 0.f;
#pragma unroll
        for (int j = 0; j < BAS; ++j) s += tmp[j] * sW2[j * INTD + c];
        float msg = xr[c] * s;
        __hip_atomic_fetch_add(&ar[c], msg, __ATOMIC_RELAXED, __HIP_MEMORY_SCOPE_AGENT);
    }
}

// ---------------------------------------------------------------------------
// Launcher
// ---------------------------------------------------------------------------
extern "C" void kernel_launch(void* const* d_in, const int* in_sizes, int n_in,
                              void* d_out, int out_size, void* d_ws, size_t ws_size,
                              hipStream_t stream) {
    (void)in_sizes; (void)n_in; (void)out_size; (void)ws_size;

    const float* m      = (const float*)d_in[0];
    const float* rbf    = (const float*)d_in[1];
    const float* sbf    = (const float*)d_in[2];
    const int*   isrc   = (const int*)d_in[3];
    const int*   idst   = (const int*)d_in[4];
    const float* Wrbf1  = (const float*)d_in[5];
    const float* Wrbf2  = (const float*)d_in[6];
    const float* Wsbf1  = (const float*)d_in[7];
    const float* Wsbf2  = (const float*)d_in[8];
    const float* Wji    = (const float*)d_in[9];
    const float* bji    = (const float*)d_in[10];
    const float* Wkj    = (const float*)d_in[11];
    const float* bkj    = (const float*)d_in[12];
    const float* Wdown  = (const float*)d_in[13];
    const float* Wup    = (const float*)d_in[14];
    const float* rbW1   = (const float*)d_in[15];
    const float* rbB1   = (const float*)d_in[16];
    const float* rbW2   = (const float*)d_in[17];
    const float* rbB2   = (const float*)d_in[18];
    const float* Wfin   = (const float*)d_in[19];
    const float* bfin   = (const float*)d_in[20];
    const float* raW1   = (const float*)d_in[21];
    const float* raB1   = (const float*)d_in[22];
    const float* raW2   = (const float*)d_in[23];
    const float* raB2   = (const float*)d_in[24];
    float* out = (float*)d_out;

    // --- workspace carve ---
    char* ws = (char*)d_ws;
    const size_t SZ128 = (size_t)EMB * EMB * sizeof(half_t);   // 32 KB
    const size_t SZ_DN = (size_t)EMB * INTD * sizeof(half_t);  // 16 KB
    size_t off = 0;
    half_t* hWji  = (half_t*)(ws + off); off += SZ128;
    half_t* hWkj  = (half_t*)(ws + off); off += SZ128;
    half_t* hWdn  = (half_t*)(ws + off); off += SZ_DN;
    half_t* hWup  = (half_t*)(ws + off); off += SZ_DN;
    half_t* hRb1  = (half_t*)(ws + off); off += SZ128;
    half_t* hRb2  = (half_t*)(ws + off); off += SZ128;
    half_t* hWfin = (half_t*)(ws + off); off += SZ128;
    half_t* hRa1a = (half_t*)(ws + off); off += SZ128;
    half_t* hRa2a = (half_t*)(ws + off); off += SZ128;
    half_t* hRa1b = (half_t*)(ws + off); off += SZ128;
    half_t* hRa2b = (half_t*)(ws + off); off += SZ128;
    float*  fWrbf = (float*)(ws + off);  off += (size_t)NRD * EMB * sizeof(float);
    off = (off + (1u << 20) - 1) & ~(size_t)((1u << 20) - 1);
    const size_t SZE128 = (size_t)E_EDGES * EMB * sizeof(float);   // 128 MB
    const size_t SZE64  = (size_t)E_EDGES * INTD * sizeof(float);  //  64 MB
    float* bufA = (float*)(ws + off); off += SZE128;   // x_ji, later residual tmp
    float* bufB = (float*)(ws + off); off += SZE128;   // m_update ping
    float* bufC = (float*)(ws + off); off += SZE128;   // x_kj gated, later pong
    float* bufD = (float*)(ws + off); off += SZE64;    // x_kj down-projected
    float* bufE = (float*)(ws + off); off += SZE64;    // agg (segment sum)

    // --- 1. weight prep ---
    auto swz = [&](const float* src, half_t* dst, int K, int N) {
        int n = K * N;
        k_swz_weights<<<(n + 255) / 256, 256, 0, stream>>>(src, dst, K, N);
    };
    swz(Wji,  hWji,  EMB, EMB);
    swz(Wkj,  hWkj,  EMB, EMB);
    swz(Wdown, hWdn, EMB, INTD);
    swz(Wup,  hWup,  INTD, EMB);
    swz(rbW1, hRb1,  EMB, EMB);
    swz(rbW2, hRb2,  EMB, EMB);
    swz(Wfin, hWfin, EMB, EMB);
    swz(raW1,             hRa1a, EMB, EMB);
    swz(raW2,             hRa2a, EMB, EMB);
    swz(raW1 + EMB * EMB, hRa1b, EMB, EMB);
    swz(raW2 + EMB * EMB, hRa2b, EMB, EMB);
    k_fuse_rbf<<<(NRD * EMB + 255) / 256, 256, 0, stream>>>(Wrbf1, Wrbf2, fWrbf);

    const dim3 g128(E_EDGES / (16 * ROWS_PER_BLK), EMB / 16);   // (2048, 8)
    const dim3 g64 (E_EDGES / (16 * ROWS_PER_BLK), INTD / 16);  // (2048, 4)

    // --- 2. edge transfer: x_ji -> bufA, gated x_kj -> bufC ---
    k_edge_transfer<<<g128, 256, 0, stream>>>(m, hWji, hWkj, bji, bkj, rbf, fWrbf,
                                              bufA, bufC);
    // --- 3. down-projection: bufD = silu(bufC @ W_down) ---
    k_gemm16<<<g64, 256, 0, stream>>>(bufC, EMB, INTD, hWdn, nullptr, nullptr, bufD);
    // --- 4. segment sum over triplets ---
    k_zero<<<2048, 256, 0, stream>>>(bufE, (size_t)E_EDGES * INTD);
    k_msg<<<T_TRIP / 256, 256, 0, stream>>>(sbf, isrc, idst, Wsbf1, Wsbf2,
                                            bufD, bufE, T_TRIP);
    // --- 5. up-projection + x_ji: bufB = silu(bufE @ W_up) + bufA ---
    k_gemm16<<<g128, 256, 0, stream>>>(bufE, INTD, EMB, hWup, nullptr, bufA, bufB);
    // --- 6. residual (before): bufB += silu(silu(bufB@W1+b1)@W2+b2) ---
    k_gemm16<<<g128, 256, 0, stream>>>(bufB, EMB, EMB, hRb1, rbB1, nullptr, bufC);
    k_gemm16<<<g128, 256, 0, stream>>>(bufC, EMB, EMB, hRb2, rbB2, bufB, bufB);
    // --- 7. final layer + skip from m: bufC = m + silu(bufB@W_final+b) ---
    k_gemm16<<<g128, 256, 0, stream>>>(bufB, EMB, EMB, hWfin, bfin, m, bufC);
    // --- 8. residual (after) x2 ---
    k_gemm16<<<g128, 256, 0, stream>>>(bufC, EMB, EMB, hRa1a, raB1, nullptr, bufA);
    k_gemm16<<<g128, 256, 0, stream>>>(bufA, EMB, EMB, hRa2a, raB2, bufC, bufB);
    k_gemm16<<<g128, 256, 0, stream>>>(bufB, EMB, EMB, hRa1b, raB1 + EMB, nullptr, bufA);
    k_gemm16<<<g128, 256, 0, stream>>>(bufA, EMB, EMB, hRa2b, raB2 + EMB, bufB, out);
}
